// VowelConsClassifier_46737834115481
// MI455X (gfx1250) — compile-verified, matrix operand order checked by reading, not run on hardware
//
#include <hip/hip_runtime.h>
#include <cstddef>

#define B_TOT   1024
#define W_IN    30
#define T_STEPS 512
#define L1      24
#define L2      20
#define C1      16
#define C2      32
#define K1      7
#define K2      5
#define KK2     80       // C1*K2 im2col depth for conv2
#define FCK     640      // C2*L2
#define BT      16       // batch tile == WMMA M
#define NTHREADS 320     // 10 waves (wave32)
#define NWAVES  10
#define TB      32       // timesteps staged per x-tile
#define TBP     33       // padded stride (bank conflicts)
#define NTB     (T_STEPS / TB)
#define S2STRIDE 648     // padded halfwords per batch row of spike2
#define ACOL_N  (L2 * 3 * 32 * 16)   // 30720 halfwords, bf16 A-fragment order
#define WCOL_N  (6 * 32 * 16)        // 3072 halfwords

typedef __attribute__((ext_vector_type(16))) __bf16 v16bf;
typedef __attribute__((ext_vector_type(8)))  float  v8f;
typedef __attribute__((ext_vector_type(2)))  float  v2f;
typedef __attribute__((ext_vector_type(4)))  unsigned int v4u;

union Frag16 { v16bf v; unsigned short u[16]; v4u q[2]; };

static __device__ __forceinline__ unsigned short f2bf(float f) {
  unsigned int u = __float_as_uint(f);
  u += 0x7FFFu + ((u >> 16) & 1u);          // round-to-nearest-even
  return (unsigned short)(u >> 16);
}
static __device__ __forceinline__ float bf2f(unsigned short b) {
  return __uint_as_float((unsigned int)b << 16);
}

__global__ __launch_bounds__(NTHREADS, 1)
void snn_vowelcons_fused(const float* __restrict__ x,
                         const float* __restrict__ w1,
                         const float* __restrict__ b1,
                         const float* __restrict__ w2,
                         const float* __restrict__ b2,
                         const float* __restrict__ fw,
                         const float* __restrict__ fb,
                         float* __restrict__ out)
{
  // +2 rows of slack: conv1 K padded 7->8 may read 1 element past w=29 (x0 weight)
  __shared__ __align__(16) float          xs[(BT * W_IN + 2) * TBP];   // 63624 B
  __shared__ __align__(16) unsigned short acol[ACOL_N];                // 61440 B spike1 im2col (A-frag order)
  __shared__ __align__(16) unsigned short wcol[WCOL_N];                // 6144 B conv2 weight B-frags
  __shared__ __align__(16) unsigned short s2s[BT * S2STRIDE];          // 20736 B spike2 (bf16 bits)
  __shared__ float fcw[2 * FCK];                                       // 5120 B
  __shared__ float part[32 * NWAVES];                                  // 1280 B

  const int tid  = threadIdx.x;
  const int lane = tid & 31;
  const int n16  = lane & 15;
  const int g    = lane >> 4;
  const int wvs  = __builtin_amdgcn_readfirstlane(tid >> 5);  // scalar wave id
  const int b0   = blockIdx.x * BT;

  // ---------------- one-time staging ----------------
  for (int i = tid; i < 2 * FCK; i += NTHREADS) fcw[i] = fw[i];

  // conv2 weight B-fragments [K=32 x N=16] bf16 in exact per-lane order
  for (int i = tid; i < WCOL_N; i += NTHREADS) {
    const int h    = i & 15;
    const int ln   = (i >> 4) & 31;
    const int f    = i >> 9;          // 0..5 = ks*2+ct
    const int ks   = f >> 1;
    const int ct   = f & 1;
    const int co   = ct * 16 + (ln & 15);
    const int kloc = (h & 7) + ((h & 8) << 1) + ((ln >> 4) << 3);
    const int kk   = ks * 32 + kloc;
    wcol[i] = (kk < KK2) ? f2bf(w2[co * KK2 + kk]) : (unsigned short)0;
  }
  // zero im2col once: padding slots (kk>=80) stay zero forever
  {
    const v4u z = {0u, 0u, 0u, 0u};
    v4u* az = (v4u*)acol;
    for (int i = tid; i < ACOL_N / 8; i += NTHREADS) az[i] = z;
  }

  // conv1 weight B-fragments (fp32, K=7 padded to 8 -> 2 k-steps of 4), registers
  v2f bw1[2];
#pragma unroll
  for (int ks = 0; ks < 2; ++ks) {
#pragma unroll
    for (int j = 0; j < 2; ++j) {
      const int k = ks * 4 + g * 2 + j;
      bw1[ks][j] = (k < K1) ? w1[n16 * K1 + k] : 0.f;
    }
  }
  const float b1reg = b1[n16];
  float cb2[2];
  cb2[0] = b2[n16];
  cb2[1] = b2[16 + n16];

  // per-(ci=n16, k) halfword offsets into an l2 fragment block of acol
  int pk[K2];
#pragma unroll
  for (int k = 0; k < K2; ++k) {
    const int kk = n16 * K2 + k;                 // 0..79
    const int ks = kk >> 5;
    const int r2 = kk & 31;
    const int g2 = (r2 >> 3) & 1;
    const int h2 = (r2 & 7) | ((r2 & 16) >> 1);
    pk[k] = ks * 512 + g2 * 256 + h2;            // + l2*1536 + b*16
  }

  // ---------------- LIF state (registers) ----------------
  float m1[3][8];                                 // lane owns (l1 slot, ci=n16, b=r+g*8)
#pragma unroll
  for (int a = 0; a < 3; ++a)
#pragma unroll
    for (int r = 0; r < 8; ++r) m1[a][r] = 0.f;
  float m2[2][2][8];                              // (li, ct, r)
#pragma unroll
  for (int a = 0; a < 2; ++a)
#pragma unroll
    for (int c = 0; c < 2; ++c)
#pragma unroll
      for (int r = 0; r < 8; ++r) m2[a][c][r] = 0.f;
  float m3 = 0.f, sum3 = 0.f;
  const float fcbias = (tid < 32) ? fb[tid & 1] : 0.f;

  for (int tb = 0; tb < NTB; ++tb) {
    const int t0 = tb * TB;
    __syncthreads();
    // stage x[b, w, t0:t0+TB] -> LDS; consecutive tid -> consecutive t (coalesced)
    for (int i = tid; i < BT * W_IN * TB; i += NTHREADS) {
      const int tl = i & (TB - 1);
      const int r  = i / TB;
      const int w  = r % W_IN;
      const int b  = r / W_IN;
      xs[(b * W_IN + w) * TBP + tl] =
          x[((size_t)(b0 + b) * W_IN + w) * T_STEPS + (t0 + tl)];
    }
    // prefetch next time-block into cache while we grind this one
    if (tb + 1 < NTB) {
      for (int i = tid; i < BT * W_IN; i += NTHREADS) {
        const int w = i % W_IN;
        const int b = i / W_IN;
        __builtin_prefetch(&x[((size_t)(b0 + b) * W_IN + w) * T_STEPS + (t0 + TB)], 0, 1);
      }
    }
    __syncthreads();

    for (int tl = 0; tl < TB; ++tl) {
      // ---- phase 1: conv1 via f32 WMMA (M=batch, N=ci, K=8) + LIF1,
      //      spikes written straight into im2col A-fragment layout ----
#pragma unroll
      for (int nl = 0; nl < 3; ++nl) {
        if (nl < 2 || wvs < 4) {                   // scalar (wave-uniform) guard
          const int l1 = (nl < 2) ? (wvs * 2 + nl) : (20 + wvs);
          const float* xb = &xs[(n16 * W_IN + l1) * TBP + tl];
          const int o0 = g * 2;
          v2f a0, a1;
          a0[0] = xb[(o0    ) * TBP];
          a0[1] = xb[(o0 + 1) * TBP];
          a1[0] = xb[(o0 + 4) * TBP];
          a1[1] = xb[(o0 + 5) * TBP];
          v8f acc = {0.f, 0.f, 0.f, 0.f, 0.f, 0.f, 0.f, 0.f};
          acc = __builtin_amdgcn_wmma_f32_16x16x4_f32(false, a0, false, bw1[0], (short)0, acc, false, false);
          acc = __builtin_amdgcn_wmma_f32_16x16x4_f32(false, a1, false, bw1[1], (short)0, acc, false, false);
          unsigned short sp[8];
#pragma unroll
          for (int r = 0; r < 8; ++r) {
            const float c1 = acc[r] + b1reg;
            const float mp = m1[nl][r];
            const float mn = __fmaf_rn(0.9f, mp, c1) - ((mp > 1.f) ? 1.f : 0.f);
            m1[nl][r] = mn;
            sp[r] = (mn > 1.f) ? (unsigned short)0x3F80 : (unsigned short)0;
          }
#pragma unroll
          for (int k = 0; k < K2; ++k) {
            const int l2 = l1 - k;
            if (l2 >= 0 && l2 < L2) {              // scalar guard
              unsigned short* ap = &acol[l2 * 1536 + pk[k] + (g << 7)];
#pragma unroll
              for (int r = 0; r < 8; ++r) ap[r * 16] = sp[r];
            }
          }
        }
      }
      __syncthreads();

      // ---- phase 2: conv2 = 20 GEMMs [16x32 += 16x96 * 96x32] via bf16 WMMA ----
      {
        Frag16 wf[3][2];
#pragma unroll
        for (int ks = 0; ks < 3; ++ks)
#pragma unroll
          for (int ct = 0; ct < 2; ++ct) {
            const v4u* p = (const v4u*)&wcol[((ks * 2 + ct) * 32 + lane) * 16];
            wf[ks][ct].q[0] = p[0];
            wf[ks][ct].q[1] = p[1];
          }
#pragma unroll
        for (int li = 0; li < 2; ++li) {
          const int l2 = wvs * 2 + li;
          Frag16 af[3];
#pragma unroll
          for (int ks = 0; ks < 3; ++ks) {
            const v4u* p = (const v4u*)&acol[((l2 * 3 + ks) * 32 + lane) * 16];
            af[ks].q[0] = p[0];
            af[ks].q[1] = p[1];
          }
#pragma unroll
          for (int ct = 0; ct < 2; ++ct) {
            v8f acc = {0.f, 0.f, 0.f, 0.f, 0.f, 0.f, 0.f, 0.f};
            acc = __builtin_amdgcn_wmma_f32_16x16x32_bf16(false, af[0].v, false, wf[0][ct].v, (short)0, acc, false, false);
            acc = __builtin_amdgcn_wmma_f32_16x16x32_bf16(false, af[1].v, false, wf[1][ct].v, (short)0, acc, false, false);
            acc = __builtin_amdgcn_wmma_f32_16x16x32_bf16(false, af[2].v, false, wf[2][ct].v, (short)0, acc, false, false);
            const int co = ct * 16 + n16;
#pragma unroll
            for (int r = 0; r < 8; ++r) {          // D: lane -> N=co, VGPR r (+g*8) -> M=b
              const int b    = r + g * 8;
              const float c2 = acc[r] + cb2[ct];
              const float mp = m2[li][ct][r];
              const float mn = __fmaf_rn(0.9f, mp, c2) - ((mp > 1.f) ? 1.f : 0.f);
              m2[li][ct][r] = mn;
              s2s[b * S2STRIDE + co * L2 + l2] =
                  (mn > 1.f) ? (unsigned short)0x3F80 : (unsigned short)0;
            }
          }
        }
      }
      __syncthreads();

      // ---- phase 3a: fc partials (32 outputs x 10 K-segments of 64) ----
      {
        const int oidx = tid / NWAVES;             // 0..31 -> (batch, out)
        const int seg  = tid - oidx * NWAVES;
        const int b    = oidx >> 1;
        const int o    = oidx & 1;
        const unsigned short* sp = &s2s[b * S2STRIDE + seg * 64];
        const float*          wp = &fcw[o * FCK + seg * 64];
        float p = 0.f;
#pragma unroll 8
        for (int k = 0; k < 64; ++k) p = __fmaf_rn(bf2f(sp[k]), wp[k], p);
        part[oidx * NWAVES + seg] = p;
      }
      __syncthreads();

      // ---- phase 3b: reduce + LIF3 + temporal mean accumulation ----
      if (tid < 32) {
        float c3 = fcbias;
#pragma unroll
        for (int s = 0; s < NWAVES; ++s) c3 += part[tid * NWAVES + s];
        const float mp = m3;
        m3 = __fmaf_rn(0.9f, mp, c3) - ((mp > 1.f) ? 1.f : 0.f);
        sum3 += m3;
      }
    }
  }

  if (tid < 32) {
    out[(size_t)(b0 + (tid >> 1)) * 2 + (tid & 1)] = sum3 * (1.0f / T_STEPS);
  }
}

extern "C" void kernel_launch(void* const* d_in, const int* in_sizes, int n_in,
                              void* d_out, int out_size, void* d_ws, size_t ws_size,
                              hipStream_t stream) {
  (void)in_sizes; (void)n_in; (void)d_ws; (void)ws_size; (void)out_size;
  const float* x  = (const float*)d_in[0];
  const float* w1 = (const float*)d_in[1];
  const float* b1 = (const float*)d_in[2];
  const float* w2 = (const float*)d_in[3];
  const float* b2 = (const float*)d_in[4];
  const float* fw = (const float*)d_in[5];
  const float* fb = (const float*)d_in[6];
  float* outp = (float*)d_out;
  snn_vowelcons_fused<<<dim3(B_TOT / BT), dim3(NTHREADS), 0, stream>>>(
      x, w1, b1, w2, b2, fw, fb, outp);
}
